// FlashMultiHeadAttention_84997402788178
// MI455X (gfx1250) — matrix operand order repaired and checked by reading, not compiled, not run on hardware
//
#include <hip/hip_runtime.h>

typedef __attribute__((ext_vector_type(16))) _Float16 v16h;
typedef __attribute__((ext_vector_type(8)))  _Float16 v8h;
typedef __attribute__((ext_vector_type(4)))  _Float16 v4h;
typedef __attribute__((ext_vector_type(8)))  float    v8f;

#define D_MODEL 1024
#define NHEADS  16
#define HDIM    64
#define BATCH   2
#define SEQ     2048
#define MROWS   (BATCH * SEQ)   // 4096

// ---------------------------------------------------------------------------
// WMMA helper: D = A(16x32 f16) * B(32x16 f16) + C(16x16 f32)
// ---------------------------------------------------------------------------
__device__ __forceinline__ v8f wmma16x16x32(v16h a, v16h b, v8f c) {
  return __builtin_amdgcn_wmma_f32_16x16x32_f16(
      /*neg_a=*/false, a, /*neg_b=*/false, b,
      /*c_mod=*/(short)0, c, /*reuse_a=*/false, /*reuse_b=*/false);
}

// A-fragment load (16x32, f16). Per ISA 7.12.2: lane L holds row M=L&15;
// K values = kb + {0..7} (VGPR0..3) and kb + {16..23} (VGPR4..7), kb = (L>=16)*8.
// rowbase points at the start of this lane's row (16B aligned).
__device__ __forceinline__ v16h frag_ld_a(const _Float16* rowbase, int kb) {
  v8h lo = *reinterpret_cast<const v8h*>(rowbase + kb);
  v8h hi = *reinterpret_cast<const v8h*>(rowbase + kb + 16);
  v16h r;
#pragma unroll
  for (int i = 0; i < 8; ++i) { r[i] = lo[i]; r[8 + i] = hi[i]; }
  return r;
}

// B-fragment load (32x16, f16). Lane L holds column N=L&15;
// K values = kb + {0..7} and kb + {16..23}, kb = (L>=16)*8 (matches A).
// rowbase points at the start of this lane's N-column source row ([N][K] tile).
__device__ __forceinline__ v16h frag_ld_b(const _Float16* rowbase, int lane) {
  const _Float16* p = rowbase + ((lane >> 4) << 3);
  v8h lo = *reinterpret_cast<const v8h*>(p);
  v8h hi = *reinterpret_cast<const v8h*>(p + 16);
  v16h r;
#pragma unroll
  for (int i = 0; i < 8; ++i) { r[i] = lo[i]; r[8 + i] = hi[i]; }
  return r;
}

// ---------------------------------------------------------------------------
// Kernel 1: QKV projection.  qkv[m,o] = sum_k x[m,k]*w_qkv[o,k] + b_qkv[o]
// M=4096, N=3072, K=1024.  Output scattered to f16 Q/K/V in [B,H,T,Dh].
// Workgroup: 256 thr = 8 waves; tile 128(M) x 128(N); wave tile 32x64.
// ---------------------------------------------------------------------------
__global__ __launch_bounds__(256) void fmha_qkv_gemm(
    const float* __restrict__ X, const float* __restrict__ W,
    const float* __restrict__ bias,
    _Float16* __restrict__ Qd, _Float16* __restrict__ Kd,
    _Float16* __restrict__ Vd) {
  __shared__ _Float16 As[128][40];  // [m][k] tile, f16, padded pitch
  __shared__ _Float16 Bs[128][40];  // [n][k] tile

  const int tid   = threadIdx.x;
  const int lane  = tid & 31;
  const int wave  = tid >> 5;
  const int waveM = wave >> 1;            // 0..3
  const int waveN = wave & 1;             // 0..1
  const int m0    = blockIdx.y * 128;
  const int n0    = blockIdx.x * 128;
  const int rl    = lane & 15;
  const int kb    = (lane >> 4) << 3;

  v8f acc[2][4] = {};

  for (int ks = 0; ks < 32; ++ks) {
    const int k0 = ks << 5;
    __syncthreads();
    // Stage A & B tiles (128x32 f32 -> f16), coalesced float4 loads.
#pragma unroll
    for (int i = 0; i < 4; ++i) {
      int e   = tid + (i << 8);           // 0..1023 float4 slots
      int row = e >> 3;
      int c4  = (e & 7) << 2;
      const float4 va = *reinterpret_cast<const float4*>(
          X + (size_t)(m0 + row) * D_MODEL + k0 + c4);
      v4h ha = {(_Float16)va.x, (_Float16)va.y, (_Float16)va.z, (_Float16)va.w};
      *reinterpret_cast<v4h*>(&As[row][c4]) = ha;
      const float4 vb = *reinterpret_cast<const float4*>(
          W + (size_t)(n0 + row) * D_MODEL + k0 + c4);
      v4h hb = {(_Float16)vb.x, (_Float16)vb.y, (_Float16)vb.z, (_Float16)vb.w};
      *reinterpret_cast<v4h*>(&Bs[row][c4]) = hb;
    }
    if (ks + 1 < 32) {  // global_prefetch_b8 for next K-slab
      __builtin_prefetch(X + (size_t)(m0 + (tid >> 1)) * D_MODEL + k0 + 32, 0, 0);
      __builtin_prefetch(W + (size_t)(n0 + (tid >> 1)) * D_MODEL + k0 + 32, 0, 0);
    }
    __syncthreads();

    v16h a[2], b[4];
#pragma unroll
    for (int i = 0; i < 2; ++i)
      a[i] = frag_ld_a(&As[waveM * 32 + i * 16 + rl][0], kb);
#pragma unroll
    for (int j = 0; j < 4; ++j)
      b[j] = frag_ld_b(&Bs[waveN * 64 + j * 16 + rl][0], lane);
#pragma unroll
    for (int i = 0; i < 2; ++i)
#pragma unroll
      for (int j = 0; j < 4; ++j)
        acc[i][j] = wmma16x16x32(a[i], b[j], acc[i][j]);
  }

  // Epilogue: add bias, scatter to Q/K/V [B,H,T,Dh] as f16.
#pragma unroll
  for (int j = 0; j < 4; ++j) {
    const int o   = n0 + waveN * 64 + j * 16 + rl;
    const float bo = bias[o];
    const int sel = o >> 10;            // 0=Q 1=K 2=V
    const int rem = o & 1023;
    const int hh  = rem >> 6;
    const int dh  = rem & 63;
    _Float16* dst = (sel == 0) ? Qd : ((sel == 1) ? Kd : Vd);
#pragma unroll
    for (int i = 0; i < 2; ++i)
#pragma unroll
      for (int rr = 0; rr < 8; ++rr) {
        const int m  = m0 + waveM * 32 + i * 16 + rr + ((lane >> 4) << 3);
        const int bb = m >> 11;
        const int tt = m & 2047;
        dst[(((size_t)bb * NHEADS + hh) * SEQ + tt) * HDIM + dh] =
            (_Float16)(acc[i][j][rr] + bo);
      }
  }
}

// ---------------------------------------------------------------------------
// Kernel 2: flash attention.  One workgroup per (b*h, 128 query rows);
// 8 waves, each owns a 16-row Q tile.  KV streamed in 32-row tiles through
// LDS (K as [kv][d], V transposed to [d][kv]).  Online softmax; P goes
// through a per-wave LDS buffer to convert C-layout -> A-layout.
// ---------------------------------------------------------------------------
__global__ __launch_bounds__(256) void fmha_flash_attn(
    const _Float16* __restrict__ Q, const _Float16* __restrict__ K,
    const _Float16* __restrict__ V, _Float16* __restrict__ O) {
  __shared__ _Float16 Ks[32][72];       // [kv][d]
  __shared__ _Float16 Vt[64][40];       // [d][kv]
  __shared__ _Float16 Ps[8][16][40];    // per-wave P tile [row][kv]

  const int tid  = threadIdx.x;
  const int lane = tid & 31;
  const int wave = tid >> 5;
  const int rl   = lane & 15;
  const int kb   = (lane >> 4) << 3;
  const int bh   = blockIdx.y;
  const int q0   = blockIdx.x * 128 + wave * 16;

  const _Float16* Qb = Q + (size_t)bh * SEQ * HDIM;
  const _Float16* Kb = K + (size_t)bh * SEQ * HDIM;
  const _Float16* Vb = V + (size_t)bh * SEQ * HDIM;

  // Q fragments, 16x64 = two 16x32 A-frags, resident for the whole kernel.
  v16h aq[2];
#pragma unroll
  for (int s = 0; s < 2; ++s)
    aq[s] = frag_ld_a(Qb + (size_t)(q0 + rl) * HDIM + s * 32, kb);

  float m_run[8], l_run[8];
#pragma unroll
  for (int r = 0; r < 8; ++r) { m_run[r] = -__builtin_inff(); l_run[r] = 0.f; }
  v8f o_acc[4] = {};

  for (int kv0 = 0; kv0 < SEQ; kv0 += 32) {
    __syncthreads();
    {  // cooperative KV staging: 256 thr x 16B, V stored transposed
      int row = tid >> 3;               // 0..31 kv row
      int c8  = (tid & 7) << 3;         // d offset
      v8h kk = *reinterpret_cast<const v8h*>(Kb + (size_t)(kv0 + row) * HDIM + c8);
      *reinterpret_cast<v8h*>(&Ks[row][c8]) = kk;
      v8h vv = *reinterpret_cast<const v8h*>(Vb + (size_t)(kv0 + row) * HDIM + c8);
#pragma unroll
      for (int j = 0; j < 8; ++j) Vt[c8 + j][row] = vv[j];
    }
    __syncthreads();

    // S = Q * K^T  (16 x 32 tile = two 16x16 C frags), scaled by 1/sqrt(64)
    v8f s[2];
#pragma unroll
    for (int ns = 0; ns < 2; ++ns) {
      v8f c = {};
#pragma unroll
      for (int ksd = 0; ksd < 2; ++ksd) {
        v16h bk = frag_ld_b(&Ks[ns * 16 + rl][ksd * 32], lane);
        c = wmma16x16x32(aq[ksd], bk, c);
      }
      s[ns] = c * 0.125f;
    }

    // Online softmax: rows live in 16-lane halves -> butterfly over xor 1..8.
    float alpha[8];
#pragma unroll
    for (int r = 0; r < 8; ++r) {
      float mx = fmaxf(s[0][r], s[1][r]);
#pragma unroll
      for (int off = 1; off < 16; off <<= 1)
        mx = fmaxf(mx, __shfl_xor(mx, off, 32));
      float mnew = fmaxf(m_run[r], mx);
      alpha[r]  = __expf(m_run[r] - mnew);
      float p0 = __expf(s[0][r] - mnew);
      float p1 = __expf(s[1][r] - mnew);
      s[0][r] = p0; s[1][r] = p1;
      float rs = p0 + p1;
#pragma unroll
      for (int off = 1; off < 16; off <<= 1)
        rs += __shfl_xor(rs, off, 32);
      l_run[r] = l_run[r] * alpha[r] + rs;
      m_run[r] = mnew;
    }
#pragma unroll
    for (int f = 0; f < 4; ++f)
#pragma unroll
      for (int r = 0; r < 8; ++r) o_acc[f][r] *= alpha[r];

    // P: C-layout -> LDS (f16) -> A-layout reload.  Wave-private buffer,
    // so an explicit DS-counter wait is enough (no block barrier needed).
    const int prow_hi = (lane >> 4) << 3;
#pragma unroll
    for (int ns = 0; ns < 2; ++ns)
#pragma unroll
      for (int r = 0; r < 8; ++r)
        Ps[wave][r + prow_hi][ns * 16 + rl] = (_Float16)s[ns][r];
    asm volatile("s_wait_dscnt 0" ::: "memory");
    v16h pa = frag_ld_a(&Ps[wave][rl][0], kb);

    // O += P * V   (four 16x16 d-subtiles)
#pragma unroll
    for (int f = 0; f < 4; ++f) {
      v16h bv = frag_ld_b(&Vt[f * 16 + rl][0], lane);
      o_acc[f] = wmma16x16x32(pa, bv, o_acc[f]);
    }
  }

  // Normalize and write attention output as f16 [B, T, H*Dh] (= [M, C]).
  const int bb = bh >> 4;
  const int hh = bh & 15;
#pragma unroll
  for (int r = 0; r < 8; ++r) {
    float inv = 1.0f / l_run[r];
    int trow = q0 + r + ((lane >> 4) << 3);
#pragma unroll
    for (int f = 0; f < 4; ++f)
      O[((size_t)bb * SEQ + trow) * D_MODEL + hh * HDIM + f * 16 + rl] =
          (_Float16)(o_acc[f][r] * inv);
  }
}

// ---------------------------------------------------------------------------
// Kernel 3: output projection. out[m,o] = sum_c attn[m,c]*w_o[o,c] + b_o[o]
// M=4096, N=1024, K=1024.  A is f16 workspace, output f32.
// ---------------------------------------------------------------------------
__global__ __launch_bounds__(256) void fmha_out_gemm(
    const _Float16* __restrict__ A, const float* __restrict__ W,
    const float* __restrict__ bias, float* __restrict__ Out) {
  __shared__ _Float16 As[128][40];
  __shared__ _Float16 Bs[128][40];

  const int tid   = threadIdx.x;
  const int lane  = tid & 31;
  const int wave  = tid >> 5;
  const int waveM = wave >> 1;
  const int waveN = wave & 1;
  const int m0    = blockIdx.y * 128;
  const int n0    = blockIdx.x * 128;
  const int rl    = lane & 15;
  const int kb    = (lane >> 4) << 3;

  v8f acc[2][4] = {};

  for (int ks = 0; ks < 32; ++ks) {
    const int k0 = ks << 5;
    __syncthreads();
#pragma unroll
    for (int i = 0; i < 2; ++i) {       // A tile: 512 v8h slots
      int e   = tid + (i << 8);
      int row = e >> 2;
      int c8  = (e & 3) << 3;
      v8h ha = *reinterpret_cast<const v8h*>(
          A + (size_t)(m0 + row) * D_MODEL + k0 + c8);
      *reinterpret_cast<v8h*>(&As[row][c8]) = ha;
    }
#pragma unroll
    for (int i = 0; i < 4; ++i) {       // B tile: f32 -> f16
      int e   = tid + (i << 8);
      int row = e >> 3;
      int c4  = (e & 7) << 2;
      const float4 vb = *reinterpret_cast<const float4*>(
          W + (size_t)(n0 + row) * D_MODEL + k0 + c4);
      v4h hb = {(_Float16)vb.x, (_Float16)vb.y, (_Float16)vb.z, (_Float16)vb.w};
      *reinterpret_cast<v4h*>(&Bs[row][c4]) = hb;
    }
    if (ks + 1 < 32) {
      __builtin_prefetch(A + (size_t)(m0 + (tid >> 1)) * D_MODEL + k0 + 32, 0, 0);
      __builtin_prefetch(W + (size_t)(n0 + (tid >> 1)) * D_MODEL + k0 + 32, 0, 0);
    }
    __syncthreads();

    v16h a[2], b[4];
#pragma unroll
    for (int i = 0; i < 2; ++i)
      a[i] = frag_ld_a(&As[waveM * 32 + i * 16 + rl][0], kb);
#pragma unroll
    for (int j = 0; j < 4; ++j)
      b[j] = frag_ld_b(&Bs[waveN * 64 + j * 16 + rl][0], lane);
#pragma unroll
    for (int i = 0; i < 2; ++i)
#pragma unroll
      for (int j = 0; j < 4; ++j)
        acc[i][j] = wmma16x16x32(a[i], b[j], acc[i][j]);
  }

#pragma unroll
  for (int j = 0; j < 4; ++j) {
    const int o   = n0 + waveN * 64 + j * 16 + rl;
    const float bo = bias[o];
#pragma unroll
    for (int i = 0; i < 2; ++i)
#pragma unroll
      for (int rr = 0; rr < 8; ++rr) {
        const int m = m0 + waveM * 32 + i * 16 + rr + ((lane >> 4) << 3);
        Out[(size_t)m * D_MODEL + o] = acc[i][j][rr] + bo;
      }
  }
}

// ---------------------------------------------------------------------------
// Launcher.
// Workspace (f16): Q | K | V (each B*H*T*Dh = 4,194,304) | attn_out (M*C).
// Total = 16M halves = 32 MB.
// ---------------------------------------------------------------------------
extern "C" void kernel_launch(void* const* d_in, const int* in_sizes, int n_in,
                              void* d_out, int out_size, void* d_ws,
                              size_t ws_size, hipStream_t stream) {
  (void)in_sizes; (void)n_in; (void)out_size; (void)ws_size;
  const float* x     = (const float*)d_in[0];
  const float* w_qkv = (const float*)d_in[1];
  const float* b_qkv = (const float*)d_in[2];
  const float* w_o   = (const float*)d_in[3];
  const float* b_o   = (const float*)d_in[4];
  float* out = (float*)d_out;

  const size_t QKV_ELEMS = (size_t)BATCH * NHEADS * SEQ * HDIM;  // 4,194,304
  _Float16* ws = (_Float16*)d_ws;
  _Float16* Qh = ws;
  _Float16* Kh = ws + QKV_ELEMS;
  _Float16* Vh = ws + 2 * QKV_ELEMS;
  _Float16* Ah = ws + 3 * QKV_ELEMS;

  // QKV projection: N=3072 -> 24 tiles, M=4096 -> 32 tiles
  fmha_qkv_gemm<<<dim3(24, 32), 256, 0, stream>>>(x, w_qkv, b_qkv, Qh, Kh, Vh);
  // Attention: 16 q-blocks of 128 rows x (B*H = 32)
  fmha_flash_attn<<<dim3(16, 32), 256, 0, stream>>>(Qh, Kh, Vh, Ah);
  // Output projection: N=1024 -> 8 tiles, M=4096 -> 32 tiles
  fmha_out_gemm<<<dim3(8, 32), 256, 0, stream>>>(Ah, w_o, b_o, out);
}
